// TransformerEncoderLayer_19774029431505
// MI455X (gfx1250) — compile-verified, matrix operand order checked by reading, not run on hardware
//
#include <hip/hip_runtime.h>
#include <math.h>

// ---------------------------------------------------------------------------
// Problem constants (from the reference): B=2, S=2048, D=1024, H=16, DK=64,
// DFF=4096, eps=1e-6.
// ---------------------------------------------------------------------------
constexpr int Bc   = 2;
constexpr int Sc   = 2048;
constexpr int Dc   = 1024;
constexpr int Hc   = 16;
constexpr int DKc  = 64;
constexpr int DFFc = 4096;
constexpr long BSc = (long)Bc * Sc;       // 4096 rows

typedef __attribute__((ext_vector_type(16))) __bf16         v16bf;
typedef __attribute__((ext_vector_type(8)))  float          v8f;
typedef __attribute__((ext_vector_type(16))) unsigned short u16x16;
typedef __attribute__((ext_vector_type(8)))  unsigned short u16x8;

// f32 -> bf16 (round-to-nearest-even), stored as raw ushort bits.
__device__ __forceinline__ unsigned short f2bf(float f) {
  union { float f; unsigned int u; } c; c.f = f;
  unsigned int u = c.u;
  return (unsigned short)((u + 0x7FFFu + ((u >> 16) & 1u)) >> 16);
}

// A/B fragment for v_wmma_f32_16x16x32_bf16 (wave32):
// lane = (m or n) % 16, h = lane/16.  Per the ISA VGPR tables the fragment is
// two contiguous 8-element chunks at K = k0+8h and K = k0+16+8h.
// NOTE: keep addressing as base-pointer + offset (NOT loop-carried pointer
// phis) so clang's address-space inference emits global_load_b128, not flat.
__device__ __forceinline__ u16x16 load_frag_bf(const unsigned short* p) {
  union { u16x16 v; u16x8 h[2]; } u;
  u.h[0] = *(const u16x8*)(p);
  u.h[1] = *(const u16x8*)(p + 16);
  return u.v;
}

__device__ __forceinline__ u16x16 load_frag_f32(const float* p) {
  const float4 a0 = *(const float4*)(p);
  const float4 a1 = *(const float4*)(p + 4);
  const float4 a2 = *(const float4*)(p + 16);
  const float4 a3 = *(const float4*)(p + 20);
  u16x16 v;
  v[0]  = f2bf(a0.x); v[1]  = f2bf(a0.y); v[2]  = f2bf(a0.z); v[3]  = f2bf(a0.w);
  v[4]  = f2bf(a1.x); v[5]  = f2bf(a1.y); v[6]  = f2bf(a1.z); v[7]  = f2bf(a1.w);
  v[8]  = f2bf(a2.x); v[9]  = f2bf(a2.y); v[10] = f2bf(a2.z); v[11] = f2bf(a2.w);
  v[12] = f2bf(a3.x); v[13] = f2bf(a3.y); v[14] = f2bf(a3.z); v[15] = f2bf(a3.w);
  return v;
}

// ---------------------------------------------------------------------------
// Generic WMMA GEMM:  C[z][m][n] = sum_k A[z][m][k] * Bt[z][n][k]  (+ epilogue)
//
// Wave tile: 32(M) x 64(N) = 2x4 fragments, 8 WMMAs per K-step of 32.
// 8 waves/block arranged WVM x WVN (WVM*WVN == 8); block tile =
// (WVM*32) x (WVN*64).  ALL launches are exact multiples of the block tile,
// so there are NO bounds guards anywhere (keeps EXEC all-ones for WMMA and
// the epilogue branch-free).
//
// z decomposed as zb = z/HB, zh = z%HB with separate offsets (for [B,H,...]
// batched attention addressing).
//
// EPI: 0 = bf16(acc + bias)                      (QKV projections)
//      1 = f32 acc*alpha with int-mask -> -1e9   (attention scores)
//      2 = bf16(acc)                             (P @ V)
//      3 = f32 acc + bias + residual             (out-proj / FFN2)
//      4 = bf16(gelu_exact(acc + bias))          (FFN1)
// ---------------------------------------------------------------------------
template <bool A_F32, int EPI, int WVM, int WVN>
__global__ __launch_bounds__(256) void wmma_gemm_kernel(
    const void* __restrict__ Ap, const unsigned short* __restrict__ Bt,
    float* __restrict__ Cf, unsigned short* __restrict__ Cb,
    const float* __restrict__ bias, const float* __restrict__ res,
    const int* __restrict__ mask,
    int K, long lda, long ldb, long ldc,
    long zA1, long zA2, long zB1, long zB2, long zC1, long zC2,
    int HB, float alpha, long maskZstride, int Nmask) {
  const int z  = blockIdx.z;
  const int zb = z / HB;
  const int zh = z % HB;
  const long aOff = (long)zb * zA1 + (long)zh * zA2;
  const long bOff = (long)zb * zB1 + (long)zh * zB2;
  const long cOff = (long)zb * zC1 + (long)zh * zC2;

  const float*          Af = (const float*)Ap;
  const unsigned short* Ab = (const unsigned short*)Ap;

  const int lane  = threadIdx.x & 31;
  const int wave  = threadIdx.x >> 5;
  const int Mbase = blockIdx.y * (WVM * 32) + (wave / WVN) * 32;
  const int Nbase = blockIdx.x * (WVN * 64) + (wave % WVN) * 64;
  const int ml = lane & 15;
  const int hh = lane >> 4;

  // Precomputed per-lane fragment offsets; indexed by k0 inside the loop.
  long offA[2], offB[4];
#pragma unroll
  for (int mt = 0; mt < 2; ++mt)
    offA[mt] = aOff + (long)(Mbase + mt * 16 + ml) * lda + 8 * hh;
#pragma unroll
  for (int nt = 0; nt < 4; ++nt)
    offB[nt] = bOff + (long)(Nbase + nt * 16 + ml) * ldb + 8 * hh;

  v8f acc[2][4] = {};

  for (int k0 = 0; k0 < K; k0 += 32) {
    u16x16 afr[2], bfr[4];
#pragma unroll
    for (int mt = 0; mt < 2; ++mt) {
      if constexpr (A_F32) afr[mt] = load_frag_f32(Af + offA[mt] + k0);
      else                 afr[mt] = load_frag_bf(Ab + offA[mt] + k0);
    }
#pragma unroll
    for (int nt = 0; nt < 4; ++nt)
      bfr[nt] = load_frag_bf(Bt + offB[nt] + k0);
#pragma unroll
    for (int mt = 0; mt < 2; ++mt)
#pragma unroll
      for (int nt = 0; nt < 4; ++nt)
        acc[mt][nt] = __builtin_amdgcn_wmma_f32_16x16x32_bf16(
            false, __builtin_bit_cast(v16bf, afr[mt]),
            false, __builtin_bit_cast(v16bf, bfr[nt]),
            (short)0, acc[mt][nt], false, false);
  }

  // C/D layout: VGPR r -> M = r + 8*(lane/16), N = lane%16.
#pragma unroll
  for (int mt = 0; mt < 2; ++mt) {
#pragma unroll
    for (int nt = 0; nt < 4; ++nt) {
      const int n = Nbase + nt * 16 + ml;
      const float bv =
          (EPI == 0 || EPI == 3 || EPI == 4) ? bias[n] : 0.0f;
      const int m0 = Mbase + mt * 16 + 8 * hh;
#pragma unroll
      for (int r = 0; r < 8; ++r) {
        const int  m  = m0 + r;
        const long ci = cOff + (long)m * ldc + n;
        const float v = acc[mt][nt][r];
        if (EPI == 0) {
          Cb[ci] = f2bf(v + bv);
        } else if (EPI == 1) {
          float sv = v * alpha;
          const long mi = (long)zb * maskZstride + (long)m * Nmask + n;
          if (mask[mi] == 0) sv = -1e9f;
          Cf[ci] = sv;
        } else if (EPI == 2) {
          Cb[ci] = f2bf(v);
        } else if (EPI == 3) {
          Cf[ci] = v + bv + res[ci];
        } else {  // EPI == 4: exact GELU
          const float x = v + bv;
          const float g = 0.5f * x * (1.0f + erff(x * 0.70710678118654752f));
          Cb[ci] = f2bf(g);
        }
      }
    }
  }
}

// ---------------------------------------------------------------------------
// Elementwise f32 -> bf16, 4 elements/thread (all sizes divisible by 1024).
// ---------------------------------------------------------------------------
__global__ void f32_to_bf16_kernel(const float4* __restrict__ in,
                                   ushort4* __restrict__ out, long n4) {
  const long i = (long)blockIdx.x * blockDim.x + threadIdx.x;
  if (i >= n4) return;
  const float4 f = in[i];
  ushort4 o;
  o.x = f2bf(f.x); o.y = f2bf(f.y); o.z = f2bf(f.z); o.w = f2bf(f.w);
  out[i] = o;
}

// ---------------------------------------------------------------------------
// V [b*S+s, h*DK+dk] (bf16)  ->  Vt [((b*H+h)*DK+dk), s]  (bf16)
// so that P@V becomes an NT GEMM with K-contiguous operands.
// ---------------------------------------------------------------------------
__global__ void v_transpose_kernel(const unsigned short* __restrict__ V,
                                   unsigned short* __restrict__ Vt) {
  const long i = (long)blockIdx.x * blockDim.x + threadIdx.x;
  const long total = BSc * Dc;
  if (i >= total) return;
  const int  d  = (int)(i % Dc);
  const long rs = i / Dc;
  const int  s  = (int)(rs % Sc);
  const int  b  = (int)(rs / Sc);
  const int  h  = d / DKc;
  const int  dk = d % DKc;
  Vt[(((long)b * Hc + h) * DKc + dk) * Sc + s] = V[i];
}

// ---------------------------------------------------------------------------
// Row softmax over S=2048 (one block of 256 per row; 8 elems/thread).
// ---------------------------------------------------------------------------
__global__ __launch_bounds__(256) void softmax_kernel(float* __restrict__ attn) {
  float* row = attn + (long)blockIdx.x * Sc;
  __shared__ float red[256];
  const int tid = threadIdx.x;
  float vals[8];
  float mx = -3.4e38f;
#pragma unroll
  for (int i = 0; i < 8; ++i) {
    vals[i] = row[tid + i * 256];
    mx = fmaxf(mx, vals[i]);
  }
  red[tid] = mx; __syncthreads();
  for (int s = 128; s > 0; s >>= 1) {
    if (tid < s) red[tid] = fmaxf(red[tid], red[tid + s]);
    __syncthreads();
  }
  mx = red[0]; __syncthreads();
  float sum = 0.0f;
#pragma unroll
  for (int i = 0; i < 8; ++i) {
    vals[i] = __expf(vals[i] - mx);
    sum += vals[i];
  }
  red[tid] = sum; __syncthreads();
  for (int s = 128; s > 0; s >>= 1) {
    if (tid < s) red[tid] += red[tid + s];
    __syncthreads();
  }
  const float inv = 1.0f / red[0];
#pragma unroll
  for (int i = 0; i < 8; ++i) row[tid + i * 256] = vals[i] * inv;
}

// ---------------------------------------------------------------------------
// LayerNorm over D=1024 with torch semantics: unbiased std (ddof=1),
// y = g*(x-mean)/(std+eps) + beta.  One block of 256 per row; 4 elems/thread.
// Optionally also emits bf16 copy (for feeding the next WMMA GEMM).
// ---------------------------------------------------------------------------
__global__ __launch_bounds__(256) void layernorm_kernel(
    const float* __restrict__ in, const float* __restrict__ g,
    const float* __restrict__ be, float* __restrict__ out,
    unsigned short* __restrict__ out_bf) {
  const float* row = in + (long)blockIdx.x * Dc;
  __shared__ float red[256];
  const int tid = threadIdx.x;
  float v[4];
  float s = 0.0f, s2 = 0.0f;
#pragma unroll
  for (int i = 0; i < 4; ++i) {
    v[i] = row[tid + i * 256];
    s += v[i];
    s2 += v[i] * v[i];
  }
  red[tid] = s; __syncthreads();
  for (int t = 128; t > 0; t >>= 1) {
    if (tid < t) red[tid] += red[tid + t];
    __syncthreads();
  }
  const float mean = red[0] / (float)Dc;
  __syncthreads();
  red[tid] = s2; __syncthreads();
  for (int t = 128; t > 0; t >>= 1) {
    if (tid < t) red[tid] += red[tid + t];
    __syncthreads();
  }
  float var = (red[0] - (float)Dc * mean * mean) / (float)(Dc - 1);
  var = fmaxf(var, 0.0f);
  const float inv = 1.0f / (sqrtf(var) + 1e-6f);
  float* orow = out + (long)blockIdx.x * Dc;
#pragma unroll
  for (int i = 0; i < 4; ++i) {
    const int j = tid + i * 256;
    const float y = g[j] * (v[i] - mean) * inv + be[j];
    orow[j] = y;
    if (out_bf) out_bf[(long)blockIdx.x * Dc + j] = f2bf(y);
  }
}

// ---------------------------------------------------------------------------
// Host-side orchestration
// ---------------------------------------------------------------------------
extern "C" void kernel_launch(void* const* d_in, const int* in_sizes, int n_in,
                              void* d_out, int out_size, void* d_ws,
                              size_t ws_size, hipStream_t stream) {
  (void)in_sizes; (void)n_in; (void)out_size; (void)ws_size;

  const float* x     = (const float*)d_in[0];
  const int*   mask  = (const int*)d_in[1];
  const float* wq    = (const float*)d_in[2];
  const float* bq    = (const float*)d_in[3];
  const float* wk    = (const float*)d_in[4];
  const float* bk    = (const float*)d_in[5];
  const float* wv    = (const float*)d_in[6];
  const float* bv    = (const float*)d_in[7];
  const float* wo    = (const float*)d_in[8];
  const float* bo    = (const float*)d_in[9];
  const float* w1    = (const float*)d_in[10];
  const float* b1    = (const float*)d_in[11];
  const float* w2    = (const float*)d_in[12];
  const float* b2    = (const float*)d_in[13];
  const float* g1    = (const float*)d_in[14];
  const float* beta1 = (const float*)d_in[15];
  const float* g2    = (const float*)d_in[16];
  const float* beta2 = (const float*)d_in[17];

  float* xout = (float*)d_out;                 // [B,S,D]
  float* attn = xout + BSc * Dc;               // [B,H,S,S]

  // -------- workspace carve (bf16 stored as ushort) --------
  char* p = (char*)d_ws;
  auto take = [&](long bytes) -> void* {
    void* r = (void*)p;
    p += (bytes + 255) & ~255L;
    return r;
  };
  unsigned short* x_bf  = (unsigned short*)take(BSc * Dc * 2);
  unsigned short* wq_bf = (unsigned short*)take((long)Dc * Dc * 2);
  unsigned short* wk_bf = (unsigned short*)take((long)Dc * Dc * 2);
  unsigned short* wv_bf = (unsigned short*)take((long)Dc * Dc * 2);
  unsigned short* wo_bf = (unsigned short*)take((long)Dc * Dc * 2);
  unsigned short* w1_bf = (unsigned short*)take((long)DFFc * Dc * 2);
  unsigned short* w2_bf = (unsigned short*)take((long)Dc * DFFc * 2);
  unsigned short* Qb    = (unsigned short*)take(BSc * Dc * 2);
  unsigned short* Kb    = (unsigned short*)take(BSc * Dc * 2);
  unsigned short* Vb    = (unsigned short*)take(BSc * Dc * 2);
  unsigned short* Vt    = (unsigned short*)take(BSc * Dc * 2);
  float*          pre   = (float*)take(BSc * Dc * 4);   // reused pre-LN buffer
  float*          h1f   = (float*)take(BSc * Dc * 4);
  unsigned short* ffb   = (unsigned short*)take(BSc * DFFc * 2);
  unsigned short* attnout_bf = Qb;   // Q dead after scores
  unsigned short* h1b        = Kb;   // K dead after scores

  const dim3 blk(256);

  // -------- 1) f32 -> bf16 conversions --------
  auto conv = [&](const float* src, unsigned short* dst, long n) {
    const long n4 = n / 4;
    f32_to_bf16_kernel<<<dim3((n4 + 255) / 256), blk, 0, stream>>>(
        (const float4*)src, (ushort4*)dst, n4);
  };
  conv(x,  x_bf,  BSc * Dc);
  conv(wq, wq_bf, (long)Dc * Dc);
  conv(wk, wk_bf, (long)Dc * Dc);
  conv(wv, wv_bf, (long)Dc * Dc);
  conv(wo, wo_bf, (long)Dc * Dc);
  conv(w1, w1_bf, (long)DFFc * Dc);
  conv(w2, w2_bf, (long)Dc * DFFc);

  // Block tile for the 2x4 wave layout: 64(M) x 256(N).
  // Block tile for the 8x1 wave layout: 256(M) x 64(N).

  // -------- 2) Q/K/V projections: bf16 out, [B*S, D] --------
  {
    const dim3 grid(Dc / 256, (int)(BSc / 64), 1);
    wmma_gemm_kernel<false, 0, 2, 4><<<grid, blk, 0, stream>>>(
        x_bf, wq_bf, nullptr, Qb, bq, nullptr, nullptr,
        Dc, Dc, Dc, Dc, 0, 0, 0, 0, 0, 0, 1, 1.0f, 0, 0);
    wmma_gemm_kernel<false, 0, 2, 4><<<grid, blk, 0, stream>>>(
        x_bf, wk_bf, nullptr, Kb, bk, nullptr, nullptr,
        Dc, Dc, Dc, Dc, 0, 0, 0, 0, 0, 0, 1, 1.0f, 0, 0);
    wmma_gemm_kernel<false, 0, 2, 4><<<grid, blk, 0, stream>>>(
        x_bf, wv_bf, nullptr, Vb, bv, nullptr, nullptr,
        Dc, Dc, Dc, Dc, 0, 0, 0, 0, 0, 0, 1, 1.0f, 0, 0);
  }

  // -------- 3) transpose V for K-contiguous P@V --------
  v_transpose_kernel<<<dim3((BSc * Dc + 255) / 256), blk, 0, stream>>>(Vb, Vt);

  // -------- 4) scores = QK^T / sqrt(DK), mask -> f32 attn buffer --------
  {
    const dim3 grid(Sc / 256, Sc / 64, Bc * Hc);
    wmma_gemm_kernel<false, 1, 2, 4><<<grid, blk, 0, stream>>>(
        Qb, Kb, attn, nullptr, nullptr, nullptr, mask,
        DKc, Dc, Dc, Sc,
        /*zA1*/ (long)Sc * Dc, /*zA2*/ DKc,
        /*zB1*/ (long)Sc * Dc, /*zB2*/ DKc,
        /*zC1*/ (long)Hc * Sc * Sc, /*zC2*/ (long)Sc * Sc,
        Hc, 0.125f, (long)Sc * Sc, Sc);
  }

  // -------- 5) softmax over last dim (writes attn_weights output) --------
  softmax_kernel<<<dim3(Bc * Hc * Sc), blk, 0, stream>>>(attn);

  // -------- 6) attn_out = P @ V  (A is f32 probs, converted in-register) ---
  //             8x1 wave layout -> exact 256x64 block tile, no guards.
  {
    const dim3 grid(DKc / 64, Sc / 256, Bc * Hc);
    wmma_gemm_kernel<true, 2, 8, 1><<<grid, blk, 0, stream>>>(
        attn, Vt, nullptr, attnout_bf, nullptr, nullptr, nullptr,
        Sc, Sc, Sc, Dc,
        /*zA1*/ (long)Hc * Sc * Sc, /*zA2*/ (long)Sc * Sc,
        /*zB1*/ (long)Hc * DKc * Sc, /*zB2*/ (long)DKc * Sc,
        /*zC1*/ (long)Sc * Dc, /*zC2*/ DKc,
        Hc, 1.0f, 0, 0);
  }

  // -------- 7) out-proj + bias + residual(x) -> pre (f32) --------
  {
    const dim3 grid(Dc / 256, (int)(BSc / 64), 1);
    wmma_gemm_kernel<false, 3, 2, 4><<<grid, blk, 0, stream>>>(
        attnout_bf, wo_bf, pre, nullptr, bo, x, nullptr,
        Dc, Dc, Dc, Dc, 0, 0, 0, 0, 0, 0, 1, 1.0f, 0, 0);
  }

  // -------- 8) layernorm1 -> h1f (f32) + h1b (bf16) --------
  layernorm_kernel<<<dim3((int)BSc), blk, 0, stream>>>(pre, g1, beta1, h1f, h1b);

  // -------- 9) FFN1: gelu(h1 @ w1^T + b1) -> ffb (bf16) --------
  {
    const dim3 grid(DFFc / 256, (int)(BSc / 64), 1);
    wmma_gemm_kernel<false, 4, 2, 4><<<grid, blk, 0, stream>>>(
        h1b, w1_bf, nullptr, ffb, b1, nullptr, nullptr,
        Dc, Dc, Dc, DFFc, 0, 0, 0, 0, 0, 0, 1, 1.0f, 0, 0);
  }

  // -------- 10) FFN2 + bias + residual(h1f) -> pre (f32) --------
  {
    const dim3 grid(Dc / 256, (int)(BSc / 64), 1);
    wmma_gemm_kernel<false, 3, 2, 4><<<grid, blk, 0, stream>>>(
        ffb, w2_bf, pre, nullptr, b2, h1f, nullptr,
        DFFc, DFFc, DFFc, Dc, 0, 0, 0, 0, 0, 0, 1, 1.0f, 0, 0);
  }

  // -------- 11) layernorm2 -> x output --------
  layernorm_kernel<<<dim3((int)BSc), blk, 0, stream>>>(pre, g2, beta2, xout,
                                                       nullptr);
}
